// EpisodicMemoryModule_44933947850771
// MI455X (gfx1250) — matrix operand order repaired and checked by reading, not compiled, not run on hardware
//
#include <hip/hip_runtime.h>
#include <math.h>

// ---------------- types ----------------
typedef __attribute__((ext_vector_type(16))) __bf16 v16bf;
typedef __attribute__((ext_vector_type(8)))  __bf16 v8bf;
typedef __attribute__((ext_vector_type(8)))  float  v8f;

#define Tdim 128
#define Bdim 64
#define Ddim 1024
#define TB   8192   // T*B
#define NSCAN_BLOCKS 16

__device__ __forceinline__ float sigm(float x) { return 1.0f / (1.0f + __expf(-x)); }

// Build the 16-half-per-lane A fragment (16x32 bf16, rows striped on lanes 0..15,
// contiguous K h*16..h*16+15 per lane half, matching the B fragment).
__device__ __forceinline__ v16bf load_a_frag(const __bf16* __restrict__ arow, int kk, int half) {
    v8bf lo = *(const v8bf*)(arow + kk + half * 16);
    v8bf hi = *(const v8bf*)(arow + kk + half * 16 + 8);
    v16bf a;
#pragma unroll
    for (int i = 0; i < 8; ++i) { a[i] = lo[i]; a[i + 8] = hi[i]; }
    return a;
}

__device__ __forceinline__ float epilogue_one(
    float x, int row, int col, int N,
    const float* bias, const float* addF, const float* addR, int act)
{
    if (bias) x += bias[col];
    if (addF) x += addF[(size_t)row * N + col];
    if (addR) x += addR[(size_t)(row & 63) * N + col];
    if (act == 1) x = tanhf(x);
    else if (act == 2) x = sigm(x);
    return x;
}

// ---------------- small GEMM (tile 64x64): for M==64 cases ----------------
__global__ __launch_bounds__(128) void k_gemm_bf16(
    const __bf16* __restrict__ A, int lda,
    const __bf16* __restrict__ Bt, int ldb,
    int M, int N, int K,
    const float* __restrict__ bias,   // [N] or null
    const float* __restrict__ addF,   // [M,N] or null
    const float* __restrict__ addR,   // [64,N] added as addR[row&63][col], or null
    int act,                          // 0 none, 1 tanh, 2 sigmoid
    float* __restrict__ outF,         // [M,N] or null
    __bf16* __restrict__ outB)        // [M,N] or null
{
    const int lane = threadIdx.x & 31;
    const int wave = threadIdx.x >> 5;
    const int half = lane >> 4;
    const int l16  = lane & 15;
    const int m0 = blockIdx.y * 64 + wave * 16;
    const int n0 = blockIdx.x * 64;

    v8f acc[4] = {};
    const __bf16* arow = A + (size_t)(m0 + l16) * lda;

    for (int kk = 0; kk < K; kk += 32) {
        v16bf a = load_a_frag(arow, kk, half);
#pragma unroll
        for (int j = 0; j < 4; ++j) {
            const __bf16* bp = Bt + (size_t)(n0 + j * 16 + l16) * ldb + kk + half * 16;
            v16bf b = *(const v16bf*)bp;
            acc[j] = __builtin_amdgcn_wmma_f32_16x16x32_bf16(
                false, a, false, b, (short)0, acc[j], false, false);
        }
    }

#pragma unroll
    for (int j = 0; j < 4; ++j) {
#pragma unroll
        for (int v = 0; v < 8; ++v) {
            int row = m0 + v + half * 8;
            int col = n0 + j * 16 + l16;
            float x = epilogue_one(acc[j][v], row, col, N, bias, addF, addR, act);
            size_t o = (size_t)row * N + col;
            if (outF) outF[o] = x;
            if (outB) outB[o] = (__bf16)x;
        }
    }
}

// ---------------- big GEMM (tile 128x128): for M==8192 cases ----------------
// 4 waves in 2x2; each wave 64x64 output (16 accums); 16 b128 loads : 16 WMMAs per k-step.
__global__ __launch_bounds__(128) void k_gemm_bf16_big(
    const __bf16* __restrict__ A, int lda,
    const __bf16* __restrict__ Bt, int ldb,
    int M, int N, int K,
    const float* __restrict__ bias,
    const float* __restrict__ addF,
    const float* __restrict__ addR,
    int act,
    float* __restrict__ outF,
    __bf16* __restrict__ outB)
{
    const int lane = threadIdx.x & 31;
    const int wave = threadIdx.x >> 5;
    const int half = lane >> 4;
    const int l16  = lane & 15;
    const int wm = wave >> 1;
    const int wn = wave & 1;
    const int m0 = blockIdx.y * 128 + wm * 64;
    const int n0 = blockIdx.x * 128 + wn * 64;

    v8f acc[4][4] = {};
    const __bf16* arow[4];
#pragma unroll
    for (int r = 0; r < 4; ++r)
        arow[r] = A + (size_t)(m0 + r * 16 + l16) * lda;
    const __bf16* brow[4];
#pragma unroll
    for (int j = 0; j < 4; ++j)
        brow[j] = Bt + (size_t)(n0 + j * 16 + l16) * ldb;

    for (int kk = 0; kk < K; kk += 32) {
        v16bf af[4], bf[4];
#pragma unroll
        for (int r = 0; r < 4; ++r) af[r] = load_a_frag(arow[r], kk, half);
#pragma unroll
        for (int j = 0; j < 4; ++j) bf[j] = *(const v16bf*)(brow[j] + kk + half * 16);
#pragma unroll
        for (int r = 0; r < 4; ++r)
#pragma unroll
            for (int j = 0; j < 4; ++j)
                acc[r][j] = __builtin_amdgcn_wmma_f32_16x16x32_bf16(
                    false, af[r], false, bf[j], (short)0, acc[r][j], false, false);
    }

#pragma unroll
    for (int r = 0; r < 4; ++r) {
#pragma unroll
        for (int j = 0; j < 4; ++j) {
#pragma unroll
            for (int v = 0; v < 8; ++v) {
                int row = m0 + r * 16 + v + half * 8;
                int col = n0 + j * 16 + l16;
                float x = epilogue_one(acc[r][j][v], row, col, N, bias, addF, addR, act);
                size_t o = (size_t)row * N + col;
                if (outF) outF[o] = x;
                if (outB) outB[o] = (__bf16)x;
            }
        }
    }
}

// ---------------- GRU tile body (shared by persistent scan and mem-GRU) ----------------
// gh = hbIn[64,1024] @ Whh[3072,1024]^T for 64-col tile d0; 3 gates; optional outer gate g.
__device__ __forceinline__ void gru_tile(
    const __bf16* __restrict__ hbIn, const __bf16* __restrict__ Whh,
    const float* __restrict__ gi, const float* __restrict__ bhh,
    const float* __restrict__ hIn, const float* __restrict__ g,
    float* __restrict__ hOut, __bf16* __restrict__ hbOut, int d0)
{
    const int lane = threadIdx.x & 31;
    const int wave = threadIdx.x >> 5;
    const int half = lane >> 4;
    const int l16  = lane & 15;
    const int m0 = wave * 16;

    v8f acc[3][4] = {};
    const __bf16* arow = hbIn + (size_t)(m0 + l16) * Ddim;

    for (int kk = 0; kk < Ddim; kk += 32) {
        v16bf a = load_a_frag(arow, kk, half);
#pragma unroll
        for (int gate = 0; gate < 3; ++gate) {
#pragma unroll
            for (int j = 0; j < 4; ++j) {
                const __bf16* bp = Whh + (size_t)(gate * Ddim + d0 + j * 16 + l16) * Ddim
                                       + kk + half * 16;
                v16bf b = *(const v16bf*)bp;
                acc[gate][j] = __builtin_amdgcn_wmma_f32_16x16x32_bf16(
                    false, a, false, b, (short)0, acc[gate][j], false, false);
            }
        }
    }

#pragma unroll
    for (int j = 0; j < 4; ++j) {
#pragma unroll
        for (int v = 0; v < 8; ++v) {
            int row = m0 + v + half * 8;
            int d = d0 + j * 16 + l16;
            const float* girow = gi + (size_t)row * 3072;
            float r = sigm(girow[d] + acc[0][j][v] + bhh[d]);
            float z = sigm(girow[Ddim + d] + acc[1][j][v] + bhh[Ddim + d]);
            float n = tanhf(girow[2 * Ddim + d] + r * (acc[2][j][v] + bhh[2 * Ddim + d]));
            float ho = hIn[(size_t)row * Ddim + d];
            float hc = (1.0f - z) * n + z * ho;
            float hn = hc;
            if (g) { float gg = g[(size_t)row * Ddim + d]; hn = gg * hc + (1.0f - gg) * ho; }
            hOut[(size_t)row * Ddim + d] = hn;
            hbOut[(size_t)row * Ddim + d] = (__bf16)hn;
        }
    }
}

// Device-scope grid barrier: monotonically increasing counter; phase = #barriers so far.
__device__ __forceinline__ void grid_barrier(unsigned* bar, unsigned phase) {
    __syncthreads();
    if (threadIdx.x == 0) {
        __hip_atomic_fetch_add(bar, 1u, __ATOMIC_RELEASE, __HIP_MEMORY_SCOPE_AGENT);
        const unsigned target = phase * (unsigned)NSCAN_BLOCKS;
        while (__hip_atomic_load(bar, __ATOMIC_ACQUIRE, __HIP_MEMORY_SCOPE_AGENT) < target)
            __builtin_amdgcn_s_sleep(1);
    }
    __syncthreads();
}

// ---------------- persistent episode scan: 128 GRU steps in ONE launch ----------------
// Grid must be exactly NSCAN_BLOCKS blocks x 128 threads (all resident).
// Block b owns h columns [b*64, b*64+64). h ping-pongs 0->1->0...; ends in buffer 0.
__global__ __launch_bounds__(128) void k_gru_scan(
    const __bf16* __restrict__ Whh,   // [3072,1024] bf16
    const float* __restrict__ CIh,    // [T*64,3072] x-side gates incl. bih
    const float* __restrict__ bhh,    // [3072]
    const float* __restrict__ G,      // [T*64,1024] outer gate
    float* __restrict__ hf0, float* __restrict__ hf1,
    __bf16* __restrict__ hb0, __bf16* __restrict__ hb1,
    unsigned* __restrict__ bar)       // zeroed before launch
{
    const int d0 = blockIdx.x * 64;
    // zero owned tile of h buffer 0
    for (int i = threadIdx.x; i < 64 * 64; i += 128) {
        int r = i >> 6, dc = i & 63;
        hf0[(size_t)r * Ddim + d0 + dc] = 0.0f;
        hb0[(size_t)r * Ddim + d0 + dc] = (__bf16)0.0f;
    }
    unsigned phase = 1;
    grid_barrier(bar, phase);

    for (int t = 0; t < Tdim; ++t) {
        const __bf16* hbIn = (t & 1) ? hb1 : hb0;
        const float*  hIn  = (t & 1) ? hf1 : hf0;
        float*        hOut = (t & 1) ? hf0 : hf1;
        __bf16*       hbOut= (t & 1) ? hb0 : hb1;
        gru_tile(hbIn, Whh, CIh + (size_t)t * Bdim * 3072, bhh,
                 hIn, G + (size_t)t * Bdim * Ddim, hOut, hbOut, d0);
        grid_barrier(bar, ++phase);
    }
}

// ---------------- standalone GRU step (memory GRU) ----------------
__global__ __launch_bounds__(128) void k_gru_step(
    const __bf16* __restrict__ hbIn, const __bf16* __restrict__ Whh,
    const float* __restrict__ gi, const float* __restrict__ bhh,
    const float* __restrict__ hIn, const float* __restrict__ g,
    float* __restrict__ hOut, __bf16* __restrict__ hbOut)
{
    gru_tile(hbIn, Whh, gi, bhh, hIn, g, hOut, hbOut, blockIdx.x * 64);
}

// ---------------- elementwise helpers ----------------
__global__ void k_convert(const float* __restrict__ s, __bf16* __restrict__ d, size_t n) {
    size_t i = (size_t)blockIdx.x * blockDim.x + threadIdx.x;
    if (i < n) d[i] = (__bf16)s[i];
}
__global__ void k_copy(const float* __restrict__ s, float* __restrict__ d, size_t n) {
    size_t i = (size_t)blockIdx.x * blockDim.x + threadIdx.x;
    if (i < n) d[i] = s[i];
}
__global__ void k_zero_u32(unsigned* p) { if (threadIdx.x == 0) *p = 0u; }
// Pack selected 1024-col blocks of W1 [1024, 9216] into contiguous bf16 [1024, nblk*1024]
__global__ void k_pack_w1(const float* __restrict__ W1, __bf16* __restrict__ dst,
                          int nblk, int b0, int b1, int b2, int b3, size_t n) {
    size_t i = (size_t)blockIdx.x * blockDim.x + threadIdx.x;
    if (i >= n) return;
    int w = nblk * Ddim;
    int r = (int)(i / w);
    int rem = (int)(i % w);
    int blkslot = rem / Ddim;
    int d = rem % Ddim;
    int blk = (blkslot == 0) ? b0 : (blkslot == 1) ? b1 : (blkslot == 2) ? b2 : b3;
    dst[i] = (__bf16)W1[(size_t)r * (9 * Ddim) + (size_t)blk * Ddim + d];
}
// Xinv[row, 0:4096] = [c, c*q, |c-q|, c*Wbq]  (bf16), row = t*64+b
__global__ void k_build_xinv(const float* __restrict__ c, const float* __restrict__ q,
                             const float* __restrict__ wbq, __bf16* __restrict__ x, size_t n) {
    size_t i = (size_t)blockIdx.x * blockDim.x + threadIdx.x;
    if (i >= n) return;
    size_t row = i / Ddim; int d = (int)(i % Ddim);
    int b = (int)(row & 63);
    float cv = c[i];
    float qv = q[(size_t)b * Ddim + d];
    float wv = wbq[(size_t)b * Ddim + d];
    __bf16* o = x + row * 4096 + d;
    o[0]        = (__bf16)cv;
    o[1 * Ddim] = (__bf16)(cv * qv);
    o[2 * Ddim] = (__bf16)fabsf(cv - qv);
    o[3 * Ddim] = (__bf16)(cv * wv);
}
// Xm[row, 0:3072] = [c*m, |c-m|, c*Wbm]  (bf16)
__global__ void k_build_xm(const float* __restrict__ c, const float* __restrict__ m,
                           const float* __restrict__ wbm, __bf16* __restrict__ x, size_t n) {
    size_t i = (size_t)blockIdx.x * blockDim.x + threadIdx.x;
    if (i >= n) return;
    size_t row = i / Ddim; int d = (int)(i % Ddim);
    int b = (int)(row & 63);
    float cv = c[i];
    float mv = m[(size_t)b * Ddim + d];
    float wv = wbm[(size_t)b * Ddim + d];
    __bf16* o = x + row * 3072 + d;
    o[0]        = (__bf16)(cv * mv);
    o[1 * Ddim] = (__bf16)fabsf(cv - mv);
    o[2 * Ddim] = (__bf16)(cv * wv);
}

// ---------------- host orchestration ----------------
static inline dim3 ew_grid(size_t n) { return dim3((unsigned)((n + 255) / 256)); }

extern "C" void kernel_launch(void* const* d_in, const int* in_sizes, int n_in,
                              void* d_out, int out_size, void* d_ws, size_t ws_size,
                              hipStream_t stream) {
    const float* c       = (const float*)d_in[0];
    const float* q       = (const float*)d_in[1];
    const float* Wb_w    = (const float*)d_in[2];
    const float* W1_w    = (const float*)d_in[3];
    const float* W1_b    = (const float*)d_in[4];
    const float* W2_w    = (const float*)d_in[5];
    const float* W2_b    = (const float*)d_in[6];
    const float* mem_Wih = (const float*)d_in[7];
    const float* mem_Whh = (const float*)d_in[8];
    const float* mem_bih = (const float*)d_in[9];
    const float* mem_bhh = (const float*)d_in[10];
    const float* att_Wih = (const float*)d_in[11];
    const float* att_Whh = (const float*)d_in[12];
    const float* att_bih = (const float*)d_in[13];
    const float* att_bhh = (const float*)d_in[14];
    (void)in_sizes; (void)n_in; (void)out_size; (void)ws_size;

    char* base = (char*)d_ws;
    size_t off = 0;
    auto carve = [&](size_t bytes) -> void* {
        void* p = base + off;
        off += (bytes + 255) & ~(size_t)255;
        return p;
    };
    // bf16 buffers
    __bf16* Xinv   = (__bf16*)carve((size_t)TB * 4096 * 2);
    __bf16* Xm     = (__bf16*)carve((size_t)TB * 3072 * 2);
    __bf16* G1b    = (__bf16*)carve((size_t)TB * Ddim * 2);
    __bf16* W1invP = (__bf16*)carve((size_t)Ddim * 4096 * 2);
    __bf16* W1mP   = (__bf16*)carve((size_t)Ddim * 3072 * 2);
    __bf16* W1b1   = (__bf16*)carve((size_t)Ddim * Ddim * 2);
    __bf16* W1b2   = (__bf16*)carve((size_t)Ddim * Ddim * 2);
    __bf16* W2b    = (__bf16*)carve((size_t)Ddim * Ddim * 2);
    __bf16* Wbb    = (__bf16*)carve((size_t)Ddim * Ddim * 2);
    __bf16* aWihB  = (__bf16*)carve((size_t)3072 * Ddim * 2);
    __bf16* aWhhB  = (__bf16*)carve((size_t)3072 * Ddim * 2);
    __bf16* mWihB  = (__bf16*)carve((size_t)3072 * Ddim * 2);
    __bf16* mWhhB  = (__bf16*)carve((size_t)3072 * Ddim * 2);
    __bf16* mb[2]; mb[0] = (__bf16*)carve((size_t)Bdim * Ddim * 2);
                   mb[1] = (__bf16*)carve((size_t)Bdim * Ddim * 2);
    __bf16* hb[2]; hb[0] = (__bf16*)carve((size_t)Bdim * Ddim * 2);
                   hb[1] = (__bf16*)carve((size_t)Bdim * Ddim * 2);
    // f32 buffers
    float* P0   = (float*)carve((size_t)TB * Ddim * 4);
    float* G    = (float*)carve((size_t)TB * Ddim * 4);
    float* CIh  = (float*)carve((size_t)TB * 3072 * 4);
    float* gtmp = (float*)carve((size_t)Bdim * 3072 * 4);
    float* qw2  = (float*)carve((size_t)Bdim * Ddim * 4);
    float* Wbq  = (float*)carve((size_t)Bdim * Ddim * 4);
    float* Wbm  = (float*)carve((size_t)Bdim * Ddim * 4);
    float* mw1  = (float*)carve((size_t)Bdim * Ddim * 4);
    float* mf[2]; mf[0] = (float*)carve((size_t)Bdim * Ddim * 4);
                  mf[1] = (float*)carve((size_t)Bdim * Ddim * 4);
    float* hf[2]; hf[0] = (float*)carve((size_t)Bdim * Ddim * 4);
                  hf[1] = (float*)carve((size_t)Bdim * Ddim * 4);
    unsigned* bar = (unsigned*)carve(256);

    const size_t nBD = (size_t)Bdim * Ddim;
    const size_t nDD = (size_t)Ddim * Ddim;
    const size_t n3D = (size_t)3072 * Ddim;

    // --- setup: weight conversions / packs ---
    k_convert<<<ew_grid(nDD), 256, 0, stream>>>(Wb_w,    Wbb,   nDD);
    k_convert<<<ew_grid(nDD), 256, 0, stream>>>(W2_w,    W2b,   nDD);
    k_convert<<<ew_grid(n3D), 256, 0, stream>>>(att_Wih, aWihB, n3D);
    k_convert<<<ew_grid(n3D), 256, 0, stream>>>(att_Whh, aWhhB, n3D);
    k_convert<<<ew_grid(n3D), 256, 0, stream>>>(mem_Wih, mWihB, n3D);
    k_convert<<<ew_grid(n3D), 256, 0, stream>>>(mem_Whh, mWhhB, n3D);
    k_pack_w1<<<ew_grid((size_t)Ddim * 4096), 256, 0, stream>>>(W1_w, W1invP, 4, 0, 3, 5, 7, (size_t)Ddim * 4096);
    k_pack_w1<<<ew_grid((size_t)Ddim * 3072), 256, 0, stream>>>(W1_w, W1mP,   3, 4, 6, 8, 0, (size_t)Ddim * 3072);
    k_pack_w1<<<ew_grid(nDD), 256, 0, stream>>>(W1_w, W1b1, 1, 1, 0, 0, 0, nDD);
    k_pack_w1<<<ew_grid(nDD), 256, 0, stream>>>(W1_w, W1b2, 1, 2, 0, 0, 0, nDD);
    // m0 = q
    k_copy<<<ew_grid(nBD), 256, 0, stream>>>(q, mf[0], nBD);
    k_convert<<<ew_grid(nBD), 256, 0, stream>>>(q, mb[0], nBD);

    dim3 blk(128);
    // qw2 = q @ W1[blk2]^T ; Wbq = q @ Wb^T
    k_gemm_bf16<<<dim3(16, 1), blk, 0, stream>>>(mb[0], Ddim, W1b2, Ddim, Bdim, Ddim, Ddim,
                                                 nullptr, nullptr, nullptr, 0, qw2, nullptr);
    k_gemm_bf16<<<dim3(16, 1), blk, 0, stream>>>(mb[0], Ddim, Wbb, Ddim, Bdim, Ddim, Ddim,
                                                 nullptr, nullptr, nullptr, 0, Wbq, nullptr);
    // Xinv features, P0, CIh (big 128x128-tile GEMMs)
    k_build_xinv<<<ew_grid((size_t)TB * Ddim), 256, 0, stream>>>(c, q, Wbq, Xinv, (size_t)TB * Ddim);
    k_gemm_bf16_big<<<dim3(8, 64), blk, 0, stream>>>(Xinv, 4096, W1invP, 4096, TB, Ddim, 4096,
                                                     W1_b, nullptr, qw2, 0, P0, nullptr);
    k_gemm_bf16_big<<<dim3(24, 64), blk, 0, stream>>>(Xinv, 4096, aWihB, Ddim, TB, 3072, Ddim,
                                                      att_bih, nullptr, nullptr, 0, CIh, nullptr);

    int cur = 0;
    for (int ep = 0; ep < 2; ++ep) {
        int nxt = cur ^ 1;
        // Wbm, m@W1[blk1]^T
        k_gemm_bf16<<<dim3(16, 1), blk, 0, stream>>>(mb[cur], Ddim, Wbb, Ddim, Bdim, Ddim, Ddim,
                                                     nullptr, nullptr, nullptr, 0, Wbm, nullptr);
        k_gemm_bf16<<<dim3(16, 1), blk, 0, stream>>>(mb[cur], Ddim, W1b1, Ddim, Bdim, Ddim, Ddim,
                                                     nullptr, nullptr, nullptr, 0, mw1, nullptr);
        // Xm, then G1 = tanh(Xm@W1m^T + P0 + mw1), G = sigmoid(G1@W2^T + b2)
        k_build_xm<<<ew_grid((size_t)TB * Ddim), 256, 0, stream>>>(c, mf[cur], Wbm, Xm, (size_t)TB * Ddim);
        k_gemm_bf16_big<<<dim3(8, 64), blk, 0, stream>>>(Xm, 3072, W1mP, 3072, TB, Ddim, 3072,
                                                         nullptr, P0, mw1, 1, nullptr, G1b);
        k_gemm_bf16_big<<<dim3(8, 64), blk, 0, stream>>>(G1b, Ddim, W2b, Ddim, TB, Ddim, Ddim,
                                                         W2_b, nullptr, nullptr, 2, G, nullptr);
        // persistent 128-step scan in one launch (h ends in buffer 0)
        k_zero_u32<<<1, 32, 0, stream>>>(bar);
        k_gru_scan<<<dim3(NSCAN_BLOCKS), blk, 0, stream>>>(
            aWhhB, CIh, att_bhh, G, hf[0], hf[1], hb[0], hb[1], bar);
        // memory GRU: e = h (buffer 0)
        k_gemm_bf16<<<dim3(48, 1), blk, 0, stream>>>(hb[0], Ddim, mWihB, Ddim, Bdim, 3072, Ddim,
                                                     mem_bih, nullptr, nullptr, 0, gtmp, nullptr);
        k_gru_step<<<dim3(16), blk, 0, stream>>>(mb[cur], mWhhB, gtmp, mem_bhh,
                                                 mf[cur], nullptr, mf[nxt], mb[nxt]);
        cur = nxt;
    }
    k_copy<<<ew_grid(nBD), 256, 0, stream>>>(mf[cur], (float*)d_out, nBD);
}
